// MixQDenseMoEBlock_59425167507873
// MI455X (gfx1250) — matrix operand order repaired and checked by reading, compiled, not run on hardware
//
#include <hip/hip_runtime.h>
#include <hip/hip_bf16.h>

// Dense all-expert MoE (SwiGLU), bf16 WMMA compute / f32 accumulate.
// Compute-bound (~824 GFLOP, AI ~1890 FLOP/B) -> v_wmma_f32_16x16x32_bf16,
// double-buffered LDS pipeline with async global->LDS A-tile copies.
// Workspace: [0, T*H) bf16 x ; [T*H, T*H+T*F) bf16 h  (~42 MB total)

typedef __bf16 bf16;
typedef bf16  v16bf __attribute__((ext_vector_type(16)));
typedef bf16  v8bf  __attribute__((ext_vector_type(8)));
typedef float v8f   __attribute__((ext_vector_type(8)));
typedef int   v4i_t __attribute__((ext_vector_type(4)));

#define HID 1024
#define FFN 4096
#define TOK 4096
#define NEXP 8

#define BM 128
#define BN 64
#define BK 64
#define KP 72   // padded LDS K-pitch (elements): 144B row pitch, 16B aligned

#if __has_builtin(__builtin_amdgcn_cvt_pk_bf16_f32)
#define HAVE_CVT_PK 1
#endif
#if __has_builtin(__builtin_amdgcn_global_load_async_to_lds_b128) && \
    __has_builtin(__builtin_amdgcn_s_wait_asynccnt)
#define HAVE_ASYNC 1
#endif

union BF16Frag { v16bf v; v8bf h[2]; };
union V8BFPack { v8bf v; unsigned u[4]; };

__device__ __forceinline__ bf16 f2bf(float f) {
  unsigned u = __builtin_bit_cast(unsigned, f);
  u += 0x7FFFu + ((u >> 16) & 1u);                 // round-to-nearest-even
  unsigned short s = (unsigned short)(u >> 16);
  return __builtin_bit_cast(bf16, s);
}

__device__ __forceinline__ unsigned pk2(float a, float b) {
#ifdef HAVE_CVT_PK
  auto p = __builtin_amdgcn_cvt_pk_bf16_f32(a, b);   // v_cvt_pk_bf16_f32
  return __builtin_bit_cast(unsigned, p);
#else
  return (unsigned)__builtin_bit_cast(unsigned short, f2bf(a)) |
         ((unsigned)__builtin_bit_cast(unsigned short, f2bf(b)) << 16);
#endif
}

// 16B global->LDS copy; async (ASYNCcnt-tracked) when available
__device__ __forceinline__ void cp16(const bf16* gsrc, bf16* ldst) {
#ifdef HAVE_ASYNC
  __builtin_amdgcn_global_load_async_to_lds_b128(
      (__attribute__((address_space(1))) v4i_t*)(void*)gsrc,
      (__attribute__((address_space(3))) v4i_t*)(void*)ldst, 0, 0);
#else
  *(uint4*)ldst = *(const uint4*)gsrc;
#endif
}

__device__ __forceinline__ void async_fence() {
#ifdef HAVE_ASYNC
  __builtin_amdgcn_s_wait_asynccnt(0);
#endif
}

__device__ __forceinline__ float fast_silu(float g) {
  float sig = __builtin_amdgcn_rcpf(1.0f + __expf(-g));  // v_rcp_f32
  return g * sig;
}

__global__ __launch_bounds__(256)
void cvt_f32_to_bf16(const float* __restrict__ x, bf16* __restrict__ y, int n) {
  int i = (blockIdx.x * 256 + threadIdx.x) * 4;
  if (i + 3 < n) {
    float4 v = *(const float4*)(x + i);
    V8BFPack p;
    p.u[0] = pk2(v.x, v.y);
    p.u[1] = pk2(v.z, v.w);
    *(uint2*)(y + i) = *(uint2*)&p.u[0];
  }
}

// ---------------------------------------------------------------------------
// Phase 1: h = silu(x @ Wg) * (x @ Wu)    [T,HID] x [HID,FFN] -> [T,FFN] bf16
// ---------------------------------------------------------------------------
__global__ __launch_bounds__(256)
void moe_swiglu_gemm(const bf16* __restrict__ xb,
                     const float* __restrict__ Wg,
                     const float* __restrict__ Wu,
                     bf16* __restrict__ hout) {
  __shared__ bf16 a_lds[2][BM * KP];   // x tile, row-major [m][k], ping-pong
  __shared__ bf16 g_lds[2][BN * KP];   // Wg tile, transposed [n][k]
  __shared__ bf16 u_lds[2][BN * KP];   // Wu tile, transposed [n][k]

  const int tid  = threadIdx.x;
  const int lane = tid & 31;
  const int wave = tid >> 5;           // 8 waves
  const int wm   = wave >> 1;          // 0..3
  const int wn   = wave & 1;           // 0..1
  const int m0   = blockIdx.y * BM;
  const int n0   = blockIdx.x * BN;

  const v8f vzero = {0.f, 0.f, 0.f, 0.f, 0.f, 0.f, 0.f, 0.f};
  v8f accg[2][2], accu[2][2];
#pragma unroll
  for (int m = 0; m < 2; ++m)
#pragma unroll
    for (int n = 0; n < 2; ++n) { accg[m][n] = vzero; accu[m][n] = vzero; }

  const int aRow = tid >> 1;           // 0..127
  const int aSeg = (tid & 1) * 32;     // 0 / 32 (elements)
  const int wN2  = lane * 2;           // 0..62
  const int wKb  = (tid >> 5) * 8;     // 0..56

  float2 rg[8], ru[8];                 // weight prefetch registers

  auto stageA = [&](int k0, int buf) {
    const bf16* src = xb + (size_t)(m0 + aRow) * HID + k0 + aSeg;
    bf16* dst = &a_lds[buf][aRow * KP + aSeg];
#pragma unroll
    for (int c = 0; c < 4; ++c) cp16(src + c * 8, dst + c * 8);
  };
  auto loadW = [&](int k0) {
#pragma unroll
    for (int i = 0; i < 8; ++i) {
      rg[i] = *(const float2*)(Wg + (size_t)(k0 + wKb + i) * FFN + n0 + wN2);
      ru[i] = *(const float2*)(Wu + (size_t)(k0 + wKb + i) * FFN + n0 + wN2);
    }
  };
  auto storeW = [&](int buf) {
    V8BFPack g0, g1, u0, u1;
#pragma unroll
    for (int j = 0; j < 4; ++j) {
      g0.u[j] = pk2(rg[2 * j].x, rg[2 * j + 1].x);
      g1.u[j] = pk2(rg[2 * j].y, rg[2 * j + 1].y);
      u0.u[j] = pk2(ru[2 * j].x, ru[2 * j + 1].x);
      u1.u[j] = pk2(ru[2 * j].y, ru[2 * j + 1].y);
    }
    *(v8bf*)(&g_lds[buf][(wN2 + 0) * KP + wKb]) = g0.v;  // ds_store_b128
    *(v8bf*)(&g_lds[buf][(wN2 + 1) * KP + wKb]) = g1.v;
    *(v8bf*)(&u_lds[buf][(wN2 + 0) * KP + wKb]) = u0.v;
    *(v8bf*)(&u_lds[buf][(wN2 + 1) * KP + wKb]) = u1.v;
  };

  // prologue: prefetch tile 0
  stageA(0, 0);
  loadW(0);

  const int NT = HID / BK;
  for (int i = 0; i < NT; ++i) {
    const int cur = i & 1;
    storeW(cur);                 // convert+store weights prefetched last iter
    async_fence();               // A copies for tile i complete
    __syncthreads();             // tile i visible to all waves
    if (i + 1 < NT) {            // kick off tile i+1 while computing tile i
      stageA((i + 1) * BK, cur ^ 1);
      loadW((i + 1) * BK);
    }

#pragma unroll
    for (int kk = 0; kk < 2; ++kk) {
      const int kb = kk * 32;
      BF16Frag a[2], bg[2], bu[2];
      const int arow  = wm * 32 + (lane & 15);
      const int khalf = (lane >> 4) * 16;
#pragma unroll
      for (int m = 0; m < 2; ++m) {
        const bf16* p = &a_lds[cur][(arow + m * 16) * KP + kb + khalf];
        a[m].h[0] = *(const v8bf*)(p);
        a[m].h[1] = *(const v8bf*)(p + 8);
      }
      const int brow = wn * 32 + (lane & 15);
      const int koff = (lane >> 4) * 16;
#pragma unroll
      for (int n = 0; n < 2; ++n) {
        const bf16* pg = &g_lds[cur][(brow + n * 16) * KP + kb + koff];
        bg[n].h[0] = *(const v8bf*)(pg);
        bg[n].h[1] = *(const v8bf*)(pg + 8);
        const bf16* pu = &u_lds[cur][(brow + n * 16) * KP + kb + koff];
        bu[n].h[0] = *(const v8bf*)(pu);
        bu[n].h[1] = *(const v8bf*)(pu + 8);
      }
#pragma unroll
      for (int m = 0; m < 2; ++m)
#pragma unroll
        for (int n = 0; n < 2; ++n) {
          accg[m][n] = __builtin_amdgcn_wmma_f32_16x16x32_bf16(
              false, a[m].v, false, bg[n].v, (short)0, accg[m][n], false, false);
          accu[m][n] = __builtin_amdgcn_wmma_f32_16x16x32_bf16(
              false, a[m].v, false, bu[n].v, (short)0, accu[m][n], false, false);
        }
    }
  }

  // ---- epilogue: silu(g)*u -> bf16 h ----
  const int erow = (lane >> 4) * 8;
  const int ecol = lane & 15;
#pragma unroll
  for (int m = 0; m < 2; ++m)
#pragma unroll
    for (int n = 0; n < 2; ++n) {
      const int baseM = m0 + wm * 32 + m * 16 + erow;
      const int baseN = n0 + wn * 32 + n * 16 + ecol;
#pragma unroll
      for (int r = 0; r < 8; ++r) {
        float g = accg[m][n][r];
        float u = accu[m][n][r];
        hout[(size_t)(baseM + r) * FFN + baseN] = f2bf(fast_silu(g) * u);
      }
    }
}

// ---------------------------------------------------------------------------
// Phase 2: out (+)= h @ Wd          [T,FFN] x [FFN,HID] -> [T,HID] f32
// ---------------------------------------------------------------------------
__global__ __launch_bounds__(256)
void moe_down_gemm(const bf16* __restrict__ hb,
                   const float* __restrict__ Wd,
                   float* __restrict__ out,
                   int accumulate) {
  __shared__ bf16 a_lds[2][BM * KP];
  __shared__ bf16 w_lds[2][BN * KP];

  const int tid  = threadIdx.x;
  const int lane = tid & 31;
  const int wave = tid >> 5;
  const int wm   = wave >> 1;
  const int wn   = wave & 1;
  const int m0   = blockIdx.y * BM;
  const int n0   = blockIdx.x * BN;

  const v8f vzero = {0.f, 0.f, 0.f, 0.f, 0.f, 0.f, 0.f, 0.f};
  v8f acc[2][2];
#pragma unroll
  for (int m = 0; m < 2; ++m)
#pragma unroll
    for (int n = 0; n < 2; ++n) acc[m][n] = vzero;

  const int aRow = tid >> 1;
  const int aSeg = (tid & 1) * 32;
  const int wN2  = lane * 2;
  const int wKb  = (tid >> 5) * 8;

  float2 rw[8];

  auto stageA = [&](int k0, int buf) {
    const bf16* src = hb + (size_t)(m0 + aRow) * FFN + k0 + aSeg;
    bf16* dst = &a_lds[buf][aRow * KP + aSeg];
#pragma unroll
    for (int c = 0; c < 4; ++c) cp16(src + c * 8, dst + c * 8);
  };
  auto loadW = [&](int k0) {
#pragma unroll
    for (int i = 0; i < 8; ++i)
      rw[i] = *(const float2*)(Wd + (size_t)(k0 + wKb + i) * HID + n0 + wN2);
  };
  auto storeW = [&](int buf) {
    V8BFPack c0, c1;
#pragma unroll
    for (int j = 0; j < 4; ++j) {
      c0.u[j] = pk2(rw[2 * j].x, rw[2 * j + 1].x);
      c1.u[j] = pk2(rw[2 * j].y, rw[2 * j + 1].y);
    }
    *(v8bf*)(&w_lds[buf][(wN2 + 0) * KP + wKb]) = c0.v;
    *(v8bf*)(&w_lds[buf][(wN2 + 1) * KP + wKb]) = c1.v;
  };

  stageA(0, 0);
  loadW(0);

  const int NT = FFN / BK;
  for (int i = 0; i < NT; ++i) {
    const int cur = i & 1;
    storeW(cur);
    async_fence();
    __syncthreads();
    if (i + 1 < NT) {
      stageA((i + 1) * BK, cur ^ 1);
      loadW((i + 1) * BK);
    }

#pragma unroll
    for (int kk = 0; kk < 2; ++kk) {
      const int kb = kk * 32;
      BF16Frag a[2], b[2];
      const int arow  = wm * 32 + (lane & 15);
      const int khalf = (lane >> 4) * 16;
#pragma unroll
      for (int m = 0; m < 2; ++m) {
        const bf16* p = &a_lds[cur][(arow + m * 16) * KP + kb + khalf];
        a[m].h[0] = *(const v8bf*)(p);
        a[m].h[1] = *(const v8bf*)(p + 8);
      }
      const int brow = wn * 32 + (lane & 15);
      const int koff = (lane >> 4) * 16;
#pragma unroll
      for (int n = 0; n < 2; ++n) {
        const bf16* pw = &w_lds[cur][(brow + n * 16) * KP + kb + koff];
        b[n].h[0] = *(const v8bf*)(pw);
        b[n].h[1] = *(const v8bf*)(pw + 8);
      }
#pragma unroll
      for (int m = 0; m < 2; ++m)
#pragma unroll
        for (int n = 0; n < 2; ++n)
          acc[m][n] = __builtin_amdgcn_wmma_f32_16x16x32_bf16(
              false, a[m].v, false, b[n].v, (short)0, acc[m][n], false, false);
    }
  }

  const int erow = (lane >> 4) * 8;
  const int ecol = lane & 15;
#pragma unroll
  for (int m = 0; m < 2; ++m)
#pragma unroll
    for (int n = 0; n < 2; ++n) {
      const int baseM = m0 + wm * 32 + m * 16 + erow;
      const int baseN = n0 + wn * 32 + n * 16 + ecol;
#pragma unroll
      for (int r = 0; r < 8; ++r) {
        size_t idx = (size_t)(baseM + r) * HID + baseN;
        float prev = accumulate ? out[idx] : 0.0f;
        out[idx] = prev + acc[m][n][r];
      }
    }
}

extern "C" void kernel_launch(void* const* d_in, const int* in_sizes, int n_in,
                              void* d_out, int out_size, void* d_ws, size_t ws_size,
                              hipStream_t stream) {
  (void)in_sizes; (void)n_in; (void)out_size; (void)ws_size;
  const float* x  = (const float*)d_in[0];   // [T, HID] f32
  const float* Wg = (const float*)d_in[1];   // [E, HID, FFN] f32
  const float* Wu = (const float*)d_in[2];   // [E, HID, FFN] f32
  const float* Wd = (const float*)d_in[3];   // [E, FFN, HID] f32
  float* out = (float*)d_out;                // [T, HID] f32

  bf16* xb = (bf16*)d_ws;                    // T*HID bf16
  bf16* hb = xb + (size_t)TOK * HID;         // T*FFN bf16

  cvt_f32_to_bf16<<<(TOK * HID) / 1024, 256, 0, stream>>>(x, xb, TOK * HID);

  for (int e = 0; e < NEXP; ++e) {
    const float* wg = Wg + (size_t)e * HID * FFN;
    const float* wu = Wu + (size_t)e * HID * FFN;
    const float* wd = Wd + (size_t)e * FFN * HID;
    moe_swiglu_gemm<<<dim3(FFN / BN, TOK / BM), 256, 0, stream>>>(xb, wg, wu, hb);
    moe_down_gemm<<<dim3(HID / BN, TOK / BM), 256, 0, stream>>>(hb, wd, out, e != 0);
  }
}
